// conv_resnet_me_block_45492293599719
// MI455X (gfx1250) — matrix operand order repaired and checked by reading, not compiled, not run on hardware
//
#include <hip/hip_runtime.h>
#include <hip/hip_bf16.h>
#include <stdint.h>

#define N_PTS 100000
#define C_CH  128
#define M_PAIRS 24576
#define BN_EPS 1e-5f

typedef __attribute__((ext_vector_type(16))) __bf16 bf16x16;
typedef __attribute__((ext_vector_type(8)))  float  f32x8;
typedef __attribute__((ext_vector_type(4)))  unsigned int u32x4;
typedef __attribute__((ext_vector_type(4)))  int v4i;

union Frag16 { bf16x16 v; u32x4 q[2]; };

// LDS row stride for a 16x128 bf16 tile: 128 + 8 pad = 272B/row (16B aligned,
// 68-dword stride -> consecutive rows land on different bank groups).
#define LDS_STRIDE 136

// ---------------------------------------------------------------------------
// 16-byte global -> LDS copy via the gfx1250 async-to-LDS path (ASYNCcnt).
// Builtin signature (from compiler diagnostic): (v4i AS1*, v4i AS3*, Iint, Iint).
// uintptr_t detour handles constness + address-space conversion; AS3 pointers
// are 32-bit and the low 32 bits of a generic LDS pointer are the LDS offset.
// ---------------------------------------------------------------------------
__device__ __forceinline__ void copy16_to_lds(__bf16* lds_dst, const __bf16* gsrc)
{
#if __has_builtin(__builtin_amdgcn_global_load_async_to_lds_b128)
    __builtin_amdgcn_global_load_async_to_lds_b128(
        (__attribute__((address_space(1))) v4i*)(uintptr_t)gsrc,
        (__attribute__((address_space(3))) v4i*)(unsigned int)(uintptr_t)lds_dst,
        0, 0);
#else
    *(u32x4*)lds_dst = *(const u32x4*)gsrc;
#endif
}

__device__ __forceinline__ void wait_async_copies()
{
#if __has_builtin(__builtin_amdgcn_global_load_async_to_lds_b128)
  #if __has_builtin(__builtin_amdgcn_s_wait_asynccnt)
    __builtin_amdgcn_s_wait_asynccnt(0);
  #else
    asm volatile("s_wait_asynccnt 0" ::: "memory");
  #endif
#endif
}

// ---------------------------------------------------------------------------
// Dense GEMM: Out = A @ B (+ resid). A bf16 row-major (ld=C), staged through
// LDS once per block; Bt bf16 pre-transposed (output-channel major, ld=C).
// Block = 16 rows x 128 cols; 8 waves, wave w -> column tile w.
// ---------------------------------------------------------------------------
template<bool RESID>
__global__ __launch_bounds__(256)
void dense_gemm_bf16(const __bf16* __restrict__ A, const __bf16* __restrict__ Bt,
                     float* __restrict__ Out, const float* __restrict__ resid)
{
    __shared__ __align__(16) __bf16 As[16 * LDS_STRIDE];

    const int t    = threadIdx.x;
    const int lane = t & 31;
    const int wave = t >> 5;            // 0..7 -> column tile
    const int half = lane >> 4;
    const int l16  = lane & 15;
    const int row0 = blockIdx.x * 16;

    // Cooperative stage of the 16x128 A tile: 256 threads x 16B.
    {
        const int srow   = t >> 4;      // 0..15
        const int schunk = t & 15;      // 0..15 (8 bf16 each)
        copy16_to_lds(&As[srow * LDS_STRIDE + schunk * 8],
                      A + (size_t)(row0 + srow) * C_CH + schunk * 8);
    }
    wait_async_copies();
    __syncthreads();

    const __bf16* bcol = Bt + (size_t)(wave * 16 + l16) * C_CH;
    __builtin_prefetch(bcol, 0, 3);

    f32x8 acc = {0.f,0.f,0.f,0.f,0.f,0.f,0.f,0.f};
    #pragma unroll
    for (int kk = 0; kk < C_CH; kk += 32) {
        Frag16 a, b;
        // A k order used here (matches B): lanes<16 K=kk..kk+15, lanes>=16 K=kk+16..kk+31
        const u32x4* pa = (const u32x4*)&As[l16 * LDS_STRIDE + kk + half * 16];
        a.q[0] = pa[0];
        a.q[1] = pa[1];                 // +8 K values = +16B
        // B k order used here (matches A): lanes<16 K=kk..kk+15, lanes>=16 K=kk+16..kk+31
        const u32x4* pb = (const u32x4*)(bcol + kk + half * 16);
        b.q[0] = pb[0];
        b.q[1] = pb[1];
        acc = __builtin_amdgcn_wmma_f32_16x16x32_bf16(
                  false, a.v, false, b.v, (short)0, acc, false, false);
    }

    const size_t obase = (size_t)row0 * C_CH + (size_t)wave * 16 + l16;
    #pragma unroll
    for (int r = 0; r < 8; ++r) {
        const size_t idx = obase + (size_t)(r + half * 8) * C_CH;
        float v = acc[r];
        if (RESID) v += resid[idx];
        Out[idx] = v;
    }
}

// ---------------------------------------------------------------------------
// Gather-GEMM-scatter for the 26 neighbor offsets:
//   contrib = Apad[in_idx[k,:],:] @ W3t[k+1];  Out[out_idx[k,m],:] += contrib
// Gathered A rows staged in LDS once per block (kills 8x redundant scattered
// global traffic). Apad row N_PTS is zeros; Out row N_PTS is the discard row.
// Grid: (M/16, 26).
// ---------------------------------------------------------------------------
__global__ __launch_bounds__(256)
void gather_gemm_scatter(const __bf16* __restrict__ Apad, const __bf16* __restrict__ W3t,
                         const int* __restrict__ in_idx, const int* __restrict__ out_idx,
                         float* __restrict__ Out)
{
    __shared__ __align__(16) __bf16 As[16 * LDS_STRIDE];

    const int k    = blockIdx.y;        // 0..25
    const int mt   = blockIdx.x;        // 0..M/16-1
    const int t    = threadIdx.x;
    const int lane = t & 31;
    const int wave = t >> 5;
    const int half = lane >> 4;
    const int l16  = lane & 15;

    const int* inrow  = in_idx  + (size_t)k * M_PAIRS + mt * 16;
    const int* outrow = out_idx + (size_t)k * M_PAIRS + mt * 16;

    // Cooperative gather-stage of the 16x128 A tile.
    {
        const int srow   = t >> 4;
        const int schunk = t & 15;
        const int grow   = inrow[srow];           // row in padded activations
        copy16_to_lds(&As[srow * LDS_STRIDE + schunk * 8],
                      Apad + (size_t)grow * C_CH + schunk * 8);
    }
    wait_async_copies();
    __syncthreads();

    const __bf16* bcol = W3t + (size_t)(k + 1) * C_CH * C_CH
                             + (size_t)(wave * 16 + l16) * C_CH;
    __builtin_prefetch(bcol, 0, 3);

    f32x8 acc = {0.f,0.f,0.f,0.f,0.f,0.f,0.f,0.f};
    #pragma unroll
    for (int kk = 0; kk < C_CH; kk += 32) {
        Frag16 a, b;
        const u32x4* pa = (const u32x4*)&As[l16 * LDS_STRIDE + kk + half * 16];
        a.q[0] = pa[0];
        a.q[1] = pa[1];
        const u32x4* pb = (const u32x4*)(bcol + kk + half * 16);
        b.q[0] = pb[0];
        b.q[1] = pb[1];
        acc = __builtin_amdgcn_wmma_f32_16x16x32_bf16(
                  false, a.v, false, b.v, (short)0, acc, false, false);
    }

    const int ncol = wave * 16 + l16;
    #pragma unroll
    for (int r = 0; r < 8; ++r) {
        const int orow = outrow[r + half * 8];
        atomicAdd(Out + (size_t)orow * C_CH + ncol, acc[r]);
    }
}

// ---------------------------------------------------------------------------
// Per-channel sum / sum-of-squares (blockDim = 128 = C).
// ---------------------------------------------------------------------------
__global__ void bn_stats(const float* __restrict__ X, float* __restrict__ stats)
{
    const int c = threadIdx.x;
    float s = 0.f, ss = 0.f;
    for (int row = blockIdx.x; row < N_PTS; row += gridDim.x) {
        const float v = X[(size_t)row * C_CH + c];
        s += v; ss += v * v;
    }
    atomicAdd(&stats[c], s);
    atomicAdd(&stats[C_CH + c], ss);
}

// BN + PReLU -> bf16 activation buffer with zeroed pad row N_PTS.
__global__ void bn_prelu_to_bf16(const float* __restrict__ X, const float* __restrict__ stats,
                                 const float* __restrict__ g, const float* __restrict__ b,
                                 const float* __restrict__ a, __bf16* __restrict__ Y)
{
    const size_t i = (size_t)blockIdx.x * blockDim.x + threadIdx.x;
    if (i >= (size_t)(N_PTS + 1) * C_CH) return;
    const int c   = (int)(i & (C_CH - 1));
    const int row = (int)(i >> 7);
    if (row == N_PTS) { Y[i] = (__bf16)0.f; return; }
    const float mu  = stats[c] * (1.f / N_PTS);
    const float var = stats[C_CH + c] * (1.f / N_PTS) - mu * mu;
    const float inv = rsqrtf(var + BN_EPS);
    float v = g[c] * (X[i] - mu) * inv + b[c];
    const float al = a[0];
    v = (v >= 0.f) ? v : al * v;
    Y[i] = (__bf16)v;
}

// Final BN + PReLU in fp32 -> d_out.
__global__ void bn_prelu_f32(const float* __restrict__ X, const float* __restrict__ stats,
                             const float* __restrict__ g, const float* __restrict__ b,
                             const float* __restrict__ a, float* __restrict__ Y)
{
    const size_t i = (size_t)blockIdx.x * blockDim.x + threadIdx.x;
    if (i >= (size_t)N_PTS * C_CH) return;
    const int c = (int)(i & (C_CH - 1));
    const float mu  = stats[c] * (1.f / N_PTS);
    const float var = stats[C_CH + c] * (1.f / N_PTS) - mu * mu;
    const float inv = rsqrtf(var + BN_EPS);
    float v = g[c] * (X[i] - mu) * inv + b[c];
    const float al = a[0];
    Y[i] = (v >= 0.f) ? v : al * v;
}

// Convert fp32 weights -> bf16, transposed to output-channel-major (Bt[d,c]).
__global__ void convert_weights(const float* __restrict__ k3, const float* __restrict__ k1,
                                __bf16* __restrict__ w3t, __bf16* __restrict__ w1t)
{
    const int i = blockIdx.x * blockDim.x + threadIdx.x;
    const int tot3 = 27 * C_CH * C_CH;
    if (i < tot3) {
        const int k   = i / (C_CH * C_CH);
        const int rem = i - k * C_CH * C_CH;
        const int c = rem >> 7;
        const int d = rem & (C_CH - 1);
        w3t[(size_t)k * C_CH * C_CH + (size_t)d * C_CH + c] = (__bf16)k3[i];
    } else if (i < tot3 + C_CH * C_CH) {
        const int j = i - tot3;
        const int c = j >> 7;
        const int d = j & (C_CH - 1);
        w1t[(size_t)d * C_CH + c] = (__bf16)k1[j];
    }
}

extern "C" void kernel_launch(void* const* d_in, const int* in_sizes, int n_in,
                              void* d_out, int out_size, void* d_ws, size_t ws_size,
                              hipStream_t stream)
{
    const float* feats = (const float*)d_in[0];
    const float* k3_w  = (const float*)d_in[1];
    const float* k1_w  = (const float*)d_in[2];
    const float* bn0_g = (const float*)d_in[3];
    const float* bn0_b = (const float*)d_in[4];
    const float* bn1_g = (const float*)d_in[5];
    const float* bn1_b = (const float*)d_in[6];
    const float* bn2_g = (const float*)d_in[7];
    const float* bn2_b = (const float*)d_in[8];
    const float* a0    = (const float*)d_in[9];
    const float* a1    = (const float*)d_in[10];
    const float* a2    = (const float*)d_in[11];
    const int*   in_idx  = (const int*)d_in[12];
    const int*   out_idx = (const int*)d_in[13];

    char* ws = (char*)d_ws;
    size_t off = 0;
    auto carve = [&](size_t bytes) -> void* {
        off = (off + 255) & ~(size_t)255;
        void* p = ws + off;
        off += bytes;
        return p;
    };
    __bf16* h_act = (__bf16*)carve((size_t)(N_PTS + 1) * C_CH * sizeof(__bf16));
    float*  fbuf  = (float*) carve((size_t)(N_PTS + 1) * C_CH * sizeof(float));
    __bf16* w3t   = (__bf16*)carve((size_t)27 * C_CH * C_CH * sizeof(__bf16));
    __bf16* w1t   = (__bf16*)carve((size_t)C_CH * C_CH * sizeof(__bf16));
    float*  stats = (float*) carve((size_t)6 * C_CH * sizeof(float));

    (void)hipMemsetAsync(stats, 0, 6 * C_CH * sizeof(float), stream);

    const int wtot = 27 * C_CH * C_CH + C_CH * C_CH;
    convert_weights<<<(wtot + 255) / 256, 256, 0, stream>>>(k3_w, k1_w, w3t, w1t);

    // Stage 0: bn0 + prelu -> bf16 activations (with zero pad row)
    bn_stats<<<1024, 128, 0, stream>>>(feats, stats);
    const int apply_blocks = (int)(((size_t)(N_PTS + 1) * C_CH + 255) / 256);
    bn_prelu_to_bf16<<<apply_blocks, 256, 0, stream>>>(feats, stats, bn0_g, bn0_b, a0, h_act);

    // Stage 1: sparse conv — dense center GEMM then 26 gather-GEMM-scatters
    dense_gemm_bf16<false><<<N_PTS / 16, 256, 0, stream>>>(h_act, w3t, fbuf, nullptr);
    dim3 gg(M_PAIRS / 16, 26);
    gather_gemm_scatter<<<gg, 256, 0, stream>>>(h_act, w3t, in_idx, out_idx, fbuf);

    // Stage 2: bn1 + prelu -> bf16, then 1x1 GEMM with residual (feats)
    bn_stats<<<1024, 128, 0, stream>>>(fbuf, stats + 2 * C_CH);
    bn_prelu_to_bf16<<<apply_blocks, 256, 0, stream>>>(fbuf, stats + 2 * C_CH, bn1_g, bn1_b, a1, h_act);
    dense_gemm_bf16<true><<<N_PTS / 16, 256, 0, stream>>>(h_act, w1t, fbuf, feats);

    // Stage 3: bn2 + prelu -> d_out (fp32)
    bn_stats<<<1024, 128, 0, stream>>>(fbuf, stats + 4 * C_CH);
    const int fin_blocks = (int)(((size_t)N_PTS * C_CH + 255) / 256);
    bn_prelu_f32<<<fin_blocks, 256, 0, stream>>>(fbuf, stats + 4 * C_CH, bn2_g, bn2_b, a2, (float*)d_out);
}
